// PTB_QLSTM_77859167142228
// MI455X (gfx1250) — matrix operand order repaired and checked by reading, not compiled, not run on hardware
//
#include <hip/hip_runtime.h>

// ---------------- problem constants ----------------
#define SEQ   128
#define BSZ   64
#define NTOK  10000
#define NTOKP 10112        // 79*128: decoder rows padded so every LDS stage is uniform
#define NH    650
#define KP    672          // K padded to 21*32 for WMMA K=32 tiles
#define G4    2600         // 4*NH
#define G4P   2608         // padded to 163*16
#define NTILES_STEP 163
#define KT_X  21           // K-tiles in x part
#define KT_TOT 42          // x part + h part
#define MROWS (SEQ*BSZ)    // 8192
#define OUTDEC ((size_t)SEQ*BSZ*NTOK)

typedef _Float16 h16;
typedef __attribute__((ext_vector_type(16))) _Float16 v16h;
typedef __attribute__((ext_vector_type(8)))  _Float16 v8h;
typedef __attribute__((ext_vector_type(8)))  float    v8f;
typedef __attribute__((ext_vector_type(4)))  int      v4i;

// ---- CDNA5 async global->LDS path (guarded; falls back to sync staging) ----
#if defined(__has_builtin)
#if __has_builtin(__builtin_amdgcn_global_load_async_to_lds_b128)
#define HAVE_ASYNC_LDS 1
#endif
#endif

#define ASG __attribute__((address_space(1)))
#define ASL __attribute__((address_space(3)))

#if defined(HAVE_ASYNC_LDS)
// builtin params (from clang diagnostic): (int4 AS1*, int4 AS3*, imm offset, imm cpol)
__device__ __forceinline__ ASG v4i* to_gas(const void* p) {
    return (ASG v4i*)(ASG void*)(void*)p;
}
__device__ __forceinline__ ASL v4i* to_las(void* p) {
    return (ASL v4i*)(ASL void*)p;
}
#if __has_builtin(__builtin_amdgcn_s_wait_asynccnt)
#define WAIT_ASYNC(n) __builtin_amdgcn_s_wait_asynccnt(n)
#else
#define WAIT_ASYNC(n) asm volatile("s_wait_asynccnt %0" :: "i"(n) : "memory")
#endif
#endif

// ---------------- helpers ----------------
__device__ __forceinline__ void atomicMaxNonNeg(float* p, float v) {
    atomicMax((unsigned int*)p, __float_as_uint(v));   // valid for non-negative floats
}

__device__ __forceinline__ float sigm(float x) { return 1.f / (1.f + expf(-x)); }

// reference quantize_act (round-half-even like jnp.round via rintf)
__device__ __forceinline__ float qact(float x, float mx, float lv) {
    float s = mx + 1e-8f;
    float t = x / s * 0.5f + 0.5f;
    t = fminf(fmaxf(t, 0.f), 1.f);
    float q = rintf(t * lv) / lv;
    return (2.f * q - 1.f) * s;
}

// A fragment 16x32 f16 (ISA 7.12.2): lane<16 -> K=0..7 & 16..23; lane>=16 -> +8
__device__ __forceinline__ v16h load_frag_a(const h16* base, int ld, int m0, int k0, int lane) {
    int m  = m0 + (lane & 15);
    int kh = (lane >> 4) << 3;               // 0 or 8
    const h16* p = base + m * ld + k0 + kh;
    v8h lo = *(const v8h*)(p);               // K = kh .. kh+7
    v8h hi = *(const v8h*)(p + 16);          // K = kh+16 .. kh+23
    return __builtin_shufflevector(lo, hi, 0,1,2,3,4,5,6,7,8,9,10,11,12,13,14,15);
}

// B fragment 32x16 f16: lane<16 -> N=lane, K=0..7 & 16..23; lane>=16 -> +8
__device__ __forceinline__ v16h load_frag_b(const h16* base, int ld, int n0, int k0, int lane) {
    int n  = n0 + (lane & 15);
    int kh = (lane >> 4) << 3;               // 0 or 8
    const h16* p = base + n * ld + k0 + kh;
    v8h lo = *(const v8h*)(p);
    v8h hi = *(const v8h*)(p + 16);
    return __builtin_shufflevector(lo, hi, 0,1,2,3,4,5,6,7,8,9,10,11,12,13,14,15);
}

__device__ __forceinline__ v8f wmma_f16(v16h a, v16h b, v8f c) {
    return __builtin_amdgcn_wmma_f32_16x16x32_f16(false, a, false, b, (short)0, c, false, false);
}

// ---------------- setup kernels ----------------
__global__ void k_zero_scal(float* xmax, float* hmax, float* omax, float* wmax) {
    int t = threadIdx.x;
    if (t < SEQ)     xmax[t] = 0.f;
    if (t < SEQ + 1) hmax[t] = 0.f;
    if (t == 0) { omax[0] = 0.f; wmax[0] = 0.f; wmax[1] = 0.f; wmax[2] = 0.f; }
}

__global__ void k_init_state(const float* h0, const float* c0, float* h, float* c) {
    int t = blockIdx.x * blockDim.x + threadIdx.x;
    if (t >= BSZ * KP) return;
    int m = t / KP, k = t % KP;
    h[t] = (k < NH) ? h0[m * NH + k] : 0.f;
    if (k < NH) c[m * NH + k] = c0[m * NH + k];
}

__global__ void k_absmax(const float* __restrict__ x, int n, float* outmax) {
    __shared__ float red[256];
    float v = 0.f;
    for (int i = blockIdx.x * blockDim.x + threadIdx.x; i < n; i += gridDim.x * blockDim.x)
        v = fmaxf(v, fabsf(x[i]));
    red[threadIdx.x] = v; __syncthreads();
    for (int s = 128; s > 0; s >>= 1) {
        if ((int)threadIdx.x < s) red[threadIdx.x] = fmaxf(red[threadIdx.x], red[threadIdx.x + s]);
        __syncthreads();
    }
    if (threadIdx.x == 0) atomicMaxNonNeg(outmax, red[0]);
}

// DoReFa weight quantization -> f16, K padded to KP, rows padded to rowsp (zeros)
__global__ void k_wquant(const float* __restrict__ w, int rows, int rowsp,
                         const float* wmaxp, const int* nb, h16* __restrict__ dst) {
    int t = blockIdx.x * blockDim.x + threadIdx.x;
    if (t >= rowsp * KP) return;
    int n = t / KP, k = t % KP;
    h16 o = (h16)0.f;
    if (n < rows && k < NH) {
        float lv = (float)((1 << *nb) - 1);
        float mt = tanhf(*wmaxp);                  // max|tanh(w)| = tanh(max|w|)
        float tn = tanhf(w[n * NH + k]) / (2.f * mt) + 0.5f;
        float q  = rintf(tn * lv) / lv;
        o = (h16)(2.f * q - 1.f);
    }
    dst[t] = o;
}

// per-step max|x_t| over gathered embeddings; block s handles step s
__global__ void k_xmax(const int* __restrict__ tokens, const float* __restrict__ embW,
                       float* xmax) {
    __shared__ float red[256];
    int s = blockIdx.x;
    float v = 0.f;
    for (int i = threadIdx.x; i < BSZ * NH; i += 256) {
        int m = i / NH, k = i % NH;
        int tok = tokens[s * BSZ + m];
        v = fmaxf(v, fabsf(embW[(size_t)tok * NH + k]));
    }
    red[threadIdx.x] = v; __syncthreads();
    for (int r = 128; r > 0; r >>= 1) {
        if ((int)threadIdx.x < r) red[threadIdx.x] = fmaxf(red[threadIdx.x], red[threadIdx.x + r]);
        __syncthreads();
    }
    if (threadIdx.x == 0) xmax[s] = red[0];
}

// quantize all x_t (embedding gather) -> f16 [SEQ][BSZ][KP]
__global__ void k_qx(const int* __restrict__ tokens, const float* __restrict__ embW,
                     const float* __restrict__ xmax, const int* nb, h16* __restrict__ qx) {
    int t = blockIdx.x * blockDim.x + threadIdx.x;
    if (t >= SEQ * BSZ * KP) return;
    int s = t / (BSZ * KP), r = t % (BSZ * KP), m = r / KP, k = r % KP;
    h16 o = (h16)0.f;
    if (k < NH) {
        float lv = (float)((1 << *nb) - 1);
        int tok = tokens[s * BSZ + m];
        o = (h16)qact(embW[(size_t)tok * NH + k], xmax[s], lv);
    }
    qx[t] = o;
}

// quantize h -> f16 [BSZ][KP]
__global__ void k_hquant(const float* __restrict__ h, const float* hmaxp, const int* nb,
                         h16* __restrict__ qh) {
    int t = blockIdx.x * blockDim.x + threadIdx.x;
    if (t >= BSZ * KP) return;
    int k = t % KP;
    float lv = (float)((1 << *nb) - 1);
    qh[t] = (k < NH) ? (h16)qact(h[t], *hmaxp, lv) : (h16)0.f;
}

// ---------------- per-step GEMM (software-pipelined one K-tile ahead) ----------------
__global__ __launch_bounds__(256) void k_step_gemm(const h16* __restrict__ qx_s,
                                                   const h16* __restrict__ qh,
                                                   const h16* __restrict__ wih,
                                                   const h16* __restrict__ whh,
                                                   float* __restrict__ gates) {
    int wave = threadIdx.x >> 5, lane = threadIdx.x & 31;
    int nt = blockIdx.x * 8 + wave;
    if (nt >= NTILES_STEP) return;

    auto loadk = [&](int kt, v16h& bb, v16h* aa) {
        const h16 *ab, *wb; int k0;
        if (kt < KT_X) { ab = qx_s; wb = wih; k0 = kt * 32; }
        else           { ab = qh;   wb = whh; k0 = (kt - KT_X) * 32; }
        bb = load_frag_b(wb, KP, nt * 16, k0, lane);
#pragma unroll
        for (int mt = 0; mt < 4; ++mt) aa[mt] = load_frag_a(ab, KP, mt * 16, k0, lane);
    };

    v8f acc[4] = {};
    v16h b, a[4], bn, an[4];
    loadk(0, b, a);
#pragma unroll 2
    for (int kt = 0; kt < KT_TOT; ++kt) {
        if (kt + 1 < KT_TOT) loadk(kt + 1, bn, an);   // issue next tile's loads first
#pragma unroll
        for (int mt = 0; mt < 4; ++mt) acc[mt] = wmma_f16(a[mt], b, acc[mt]);
        b = bn;
#pragma unroll
        for (int mt = 0; mt < 4; ++mt) a[mt] = an[mt];
    }
    int n  = nt * 16 + (lane & 15);
    int mo = (lane >> 4) * 8;
#pragma unroll
    for (int mt = 0; mt < 4; ++mt)
#pragma unroll
        for (int v = 0; v < 8; ++v)
            gates[(mt * 16 + mo + v) * G4P + n] = acc[mt][v];
}

// ---------------- LSTM pointwise ----------------
__global__ void k_pointwise(const float* __restrict__ gates,
                            const float* __restrict__ bih, const float* __restrict__ bhh,
                            float* __restrict__ h, float* __restrict__ c,
                            float* __restrict__ outb, float* hmax_next, float* omax, int s) {
    __shared__ float red[256];
    int t = blockIdx.x * blockDim.x + threadIdx.x;
    float amx = 0.f;
    if (t < BSZ * KP) {
        int m = t / KP, k = t % KP;
        if (k < NH) {
            float gi = gates[m * G4P + k]            + bih[k]            + bhh[k];
            float gf = gates[m * G4P + NH + k]       + bih[NH + k]       + bhh[NH + k];
            float gg = gates[m * G4P + 2 * NH + k]   + bih[2 * NH + k]   + bhh[2 * NH + k];
            float go = gates[m * G4P + 3 * NH + k]   + bih[3 * NH + k]   + bhh[3 * NH + k];
            float cn = sigm(gf) * c[m * NH + k] + sigm(gi) * tanhf(gg);
            float hn = sigm(go) * tanhf(cn);
            c[m * NH + k] = cn;
            h[t] = hn;
            outb[(size_t)(s * BSZ + m) * KP + k] = hn;
            amx = fabsf(hn);
        } else {
            h[t] = 0.f;
            outb[(size_t)(s * BSZ + t / KP) * KP + k] = 0.f;
        }
    }
    red[threadIdx.x] = amx; __syncthreads();
    for (int r = 128; r > 0; r >>= 1) {
        if ((int)threadIdx.x < r) red[threadIdx.x] = fmaxf(red[threadIdx.x], red[threadIdx.x + r]);
        __syncthreads();
    }
    if (threadIdx.x == 0) {
        atomicMaxNonNeg(hmax_next, red[0]);
        atomicMaxNonNeg(omax, red[0]);
    }
}

// quantize the whole out history with global scale -> f16 [MROWS][KP]
__global__ void k_qo(const float* __restrict__ outb, const float* omax, const int* nb,
                     h16* __restrict__ qo) {
    int t = blockIdx.x * blockDim.x + threadIdx.x;
    if (t >= MROWS * KP) return;
    int k = t % KP;
    float lv = (float)((1 << *nb) - 1);
    qo[t] = (k < NH) ? (h16)qact(outb[t], *omax, lv) : (h16)0.f;
}

// ------ decoder GEMM: [8192 x KP] * [NTOKP x KP]^T, async double-buffered LDS ------
__global__ __launch_bounds__(256) void k_dec_gemm(const h16* __restrict__ qo,
                                                  const h16* __restrict__ wdec,
                                                  const float* __restrict__ db,
                                                  float* __restrict__ dout) {
    __shared__ h16 As[2][128 * 32];
    __shared__ h16 Bs[2][128 * 32];
    int wave = threadIdx.x >> 5, lane = threadIdx.x & 31;
    int mbase = blockIdx.x * 128;
    int nt    = blockIdx.y * 8 + wave;       // global n-tile of this wave
    int ldr = threadIdx.x >> 1;              // 0..127 loader row
    int lcs = (threadIdx.x & 1) * 16;        // 0 or 16 halves
    int nrow = blockIdx.y * 128 + ldr;       // global B row (always < NTOKP, zero-padded)

    const h16* gA = qo   + (size_t)(mbase + ldr) * KP + lcs;
    const h16* gB = wdec + (size_t)nrow * KP + lcs;
    h16* lA = &As[0][ldr * 32 + lcs];
    h16* lB = &Bs[0][ldr * 32 + lcs];

    auto stage = [&](int buf, int kt) {      // 4 x 16B per thread per stage
        int k0 = kt * 32;
        int lofs = buf * 128 * 32;
#if defined(HAVE_ASYNC_LDS)
        __builtin_amdgcn_global_load_async_to_lds_b128(to_gas(gA + k0),     to_las(lA + lofs),     0, 0);
        __builtin_amdgcn_global_load_async_to_lds_b128(to_gas(gA + k0 + 8), to_las(lA + lofs + 8), 0, 0);
        __builtin_amdgcn_global_load_async_to_lds_b128(to_gas(gB + k0),     to_las(lB + lofs),     0, 0);
        __builtin_amdgcn_global_load_async_to_lds_b128(to_gas(gB + k0 + 8), to_las(lB + lofs + 8), 0, 0);
#else
        *(uint4*)(lA + lofs)     = *(const uint4*)(gA + k0);
        *(uint4*)(lA + lofs + 8) = *(const uint4*)(gA + k0 + 8);
        *(uint4*)(lB + lofs)     = *(const uint4*)(gB + k0);
        *(uint4*)(lB + lofs + 8) = *(const uint4*)(gB + k0 + 8);
#endif
    };

    v8f acc[8] = {};
    stage(0, 0);
    for (int kt = 0; kt < 21; ++kt) {
        int cur = kt & 1;
        if (kt < 20) stage(cur ^ 1, kt + 1);      // prefetch next tile into other buffer
#if defined(HAVE_ASYNC_LDS)
        if (kt < 20) WAIT_ASYNC(4);               // async loads retire in order:
        else         WAIT_ASYNC(0);               // <=4 left => stage kt complete
#endif
        __syncthreads();
        v16h b = load_frag_b(&Bs[cur][0], 32, wave * 16, 0, lane);
#pragma unroll
        for (int mt = 0; mt < 8; ++mt) {
            v16h a = load_frag_a(&As[cur][0], 32, mt * 16, 0, lane);
            acc[mt] = wmma_f16(a, b, acc[mt]);
        }
        __syncthreads();                          // buf[cur] free for stage kt+2
    }
    if (nt < NTOK / 16) {
        int n  = nt * 16 + (lane & 15);
        float bias = db[n];
        int mo = (lane >> 4) * 8;
#pragma unroll
        for (int mt = 0; mt < 8; ++mt)
#pragma unroll
            for (int v = 0; v < 8; ++v)
                dout[(size_t)(mbase + mt * 16 + mo + v) * NTOK + n] = acc[mt][v] + bias;
    }
}

// final h/c copy into d_out tail (strip K padding)
__global__ void k_tail(const float* __restrict__ h, const float* __restrict__ c,
                       float* __restrict__ dst) {
    int t = blockIdx.x * blockDim.x + threadIdx.x;
    if (t >= 2 * BSZ * NH) return;
    if (t < BSZ * NH) { int m = t / NH, k = t % NH; dst[t] = h[m * KP + k]; }
    else              { dst[t] = c[t - BSZ * NH]; }
}

// ---------------- host ----------------
extern "C" void kernel_launch(void* const* d_in, const int* in_sizes, int n_in,
                              void* d_out, int out_size, void* d_ws, size_t ws_size,
                              hipStream_t stream) {
    const int*   tokens = (const int*)  d_in[0];
    const float* h0     = (const float*)d_in[1];
    const float* c0     = (const float*)d_in[2];
    const float* embW   = (const float*)d_in[3];
    const float* Wih    = (const float*)d_in[4];
    const float* Whh    = (const float*)d_in[5];
    const float* bih    = (const float*)d_in[6];
    const float* bhh    = (const float*)d_in[7];
    const float* Wd     = (const float*)d_in[8];
    const float* db     = (const float*)d_in[9];
    const int*   nb     = (const int*)  d_in[10];

    char* ws = (char*)d_ws;
    size_t off = 0;
    auto alloc = [&](size_t bytes) -> void* {
        void* p = ws + off;
        off = (off + bytes + 255) & ~(size_t)255;
        return p;
    };
    h16*  wq_ih = (h16*) alloc((size_t)G4P * KP * 2);
    h16*  wq_hh = (h16*) alloc((size_t)G4P * KP * 2);
    h16*  wq_dec= (h16*) alloc((size_t)NTOKP * KP * 2);
    h16*  qx    = (h16*) alloc((size_t)SEQ * BSZ * KP * 2);
    h16*  qh    = (h16*) alloc((size_t)BSZ * KP * 2);
    float* hbuf = (float*)alloc((size_t)BSZ * KP * 4);
    float* cbuf = (float*)alloc((size_t)BSZ * NH * 4);
    float* gates= (float*)alloc((size_t)BSZ * G4P * 4);
    float* outb = (float*)alloc((size_t)MROWS * KP * 4);
    h16*  qo    = (h16*) alloc((size_t)MROWS * KP * 2);
    float* xmax = (float*)alloc(SEQ * 4);
    float* hmax = (float*)alloc((SEQ + 1) * 4);
    float* omax = (float*)alloc(4);
    float* wmax = (float*)alloc(3 * 4);

    // setup
    k_zero_scal<<<1, 256, 0, stream>>>(xmax, hmax, omax, wmax);
    k_init_state<<<(BSZ * KP + 255) / 256, 256, 0, stream>>>(h0, c0, hbuf, cbuf);
    k_absmax<<<512, 256, 0, stream>>>(Wih, G4 * NH, wmax + 0);
    k_absmax<<<512, 256, 0, stream>>>(Whh, G4 * NH, wmax + 1);
    k_absmax<<<512, 256, 0, stream>>>(Wd,  NTOK * NH, wmax + 2);
    k_absmax<<<64, 256, 0, stream>>>(h0, BSZ * NH, hmax + 0);   // scale for initial h
    k_wquant<<<(G4P * KP + 255) / 256, 256, 0, stream>>>(Wih, G4, G4P, wmax + 0, nb, wq_ih);
    k_wquant<<<(G4P * KP + 255) / 256, 256, 0, stream>>>(Whh, G4, G4P, wmax + 1, nb, wq_hh);
    k_wquant<<<(NTOKP * KP + 255) / 256, 256, 0, stream>>>(Wd, NTOK, NTOKP, wmax + 2, nb, wq_dec);
    k_xmax<<<SEQ, 256, 0, stream>>>(tokens, embW, xmax);
    k_qx<<<(SEQ * BSZ * KP + 255) / 256, 256, 0, stream>>>(tokens, embW, xmax, nb, qx);
    k_hquant<<<(BSZ * KP + 255) / 256, 256, 0, stream>>>(hbuf, hmax + 0, nb, qh);

    // recurrence
    for (int s = 0; s < SEQ; ++s) {
        k_step_gemm<<<(NTILES_STEP + 7) / 8, 256, 0, stream>>>(
            qx + (size_t)s * BSZ * KP, qh, wq_ih, wq_hh, gates);
        k_pointwise<<<(BSZ * KP + 255) / 256, 256, 0, stream>>>(
            gates, bih, bhh, hbuf, cbuf, outb, hmax + s + 1, omax, s);
        k_hquant<<<(BSZ * KP + 255) / 256, 256, 0, stream>>>(hbuf, hmax + s + 1, nb, qh);
    }

    // decode
    k_qo<<<(MROWS * KP + 255) / 256, 256, 0, stream>>>(outb, omax, nb, qo);
    dim3 dgrid(MROWS / 128, NTOKP / 128);  // 64 x 79
    k_dec_gemm<<<dgrid, 256, 0, stream>>>(qo, wq_dec, db, (float*)d_out);
    k_tail<<<(2 * BSZ * NH + 255) / 256, 256, 0, stream>>>(hbuf, cbuf, (float*)d_out + OUTDEC);
}
